// LIVOperator_45853070852491
// MI455X (gfx1250) — matrix operand order repaired and checked by reading, not compiled, not run on hardware
//
#include <hip/hip_runtime.h>
#include <hip/hip_bf16.h>

typedef __attribute__((ext_vector_type(16))) __bf16 v16bf;
typedef __attribute__((ext_vector_type(8)))  __bf16 v8bf;
typedef __attribute__((ext_vector_type(8)))  float  v8f;

#define B_ 4
#define T_ 4096
#define D_ 1024
#define M_ (B_ * T_)   // 16384
#define K_ D_          // 1024
#define N_ (2 * D_)    // 2048

// GEMM block tile: 128 (M) x 64 (N), K-step 32. 256 threads = 8 waves.
// Wave grid 4x2, each wave owns a 32x32 register tile (4 WMMAs).
#define BM 128
#define BN 64
#define BK 32

#define NC 32               // scan chunks along T
#define CL (T_ / NC)        // 128 steps per chunk

__device__ __forceinline__ float sigmoidf_(float v) {
    return 1.0f / (1.0f + __expf(-v));
}

// XOR swizzle of 16B blocks within a 64B LDS row: kills the 4-way bank
// conflict among lanes r, r+4, r+8, r+12 (rows are 64B apart).
__device__ __forceinline__ int swz_blk(int row, int blk) {
    return (blk ^ ((row >> 2) & 3)) & 3;
}

// ---------------------------------------------------------------------------
// Kernel 1: gates = sigmoid(X·Wg + bg), routed to ig (n<D) / fg (n>=D).
// f32 -> bf16 at LDS staging; register-double-buffered global loads so the
// next tile's global_load_b128s are in flight across the WMMA section;
// fragment gathers are swizzled ds_load_b128.
// ---------------------------------------------------------------------------
__global__ __launch_bounds__(256) void gates_gemm_kernel(
    const float* __restrict__ X,   // [M_, K_]
    const float* __restrict__ Wg,  // [K_, N_]
    const float* __restrict__ bg,  // [N_]
    float* __restrict__ ig,        // [M_, D_]
    float* __restrict__ fg)        // [M_, D_]
{
    // A: row-major [row][k], 32 bf16 (64B) per row.
    // Bt: transposed [n][k], 32 bf16 (64B) per column-of-B.
    __shared__ __align__(16) __bf16 As[BM][BK];   // 8 KB
    __shared__ __align__(16) __bf16 Bt[BN][BK];   // 4 KB

    const int tid  = threadIdx.x;
    const int lane = tid & 31;
    const int wave = tid >> 5;

    const int bm = blockIdx.y;   // 0..127
    const int bn = blockIdx.x;   // 0..31

    const int wm = wave & 3;     // wave's 32-row tile within 128
    const int wn = wave >> 2;    // wave's 32-col tile within 64

    const int r    = lane & 15;
    const int half = lane >> 4;

    // ---- staging thread mapping (fixed per thread) ----
    // A: 4 float4 per thread covering 128 rows x 32 cols.
    const int aRow[4] = { (tid + 0) >> 3, (tid + 256) >> 3,
                          (tid + 512) >> 3, (tid + 768) >> 3 };
    const int aCol4   = (tid & 7) << 2;               // 0,4,..,28
    // B: 2 float4 per thread, two k-adjacent rows, covering 32 rows x 64 cols.
    const int bRow2 = (tid >> 4) << 1;                // 0,2,..,30
    const int bCol4 = (tid & 15) << 2;                // 0,4,..,60

    float4 aR[4];
    float4 bR[2];

    auto issue_loads = [&](int k0) {
        #pragma unroll
        for (int it = 0; it < 4; ++it)
            aR[it] = *(const float4*)&X[(size_t)(bm * BM + aRow[it]) * K_ + k0 + aCol4];
        #pragma unroll
        for (int it = 0; it < 2; ++it)
            bR[it] = *(const float4*)&Wg[(size_t)(k0 + bRow2 + it) * N_ + bn * BN + bCol4];
    };
    auto store_lds = [&]() {
        #pragma unroll
        for (int it = 0; it < 4; ++it) {
            const int row = aRow[it];
            const int cc  = swz_blk(row, aCol4 >> 3) * 8 + (aCol4 & 7);
            As[row][cc + 0] = (__bf16)aR[it].x;
            As[row][cc + 1] = (__bf16)aR[it].y;
            As[row][cc + 2] = (__bf16)aR[it].z;
            As[row][cc + 3] = (__bf16)aR[it].w;
        }
        // Transpose-scatter B: pack (k, k+1) bf16 pairs -> one ds_store_b32.
        const float v0[4] = {bR[0].x, bR[0].y, bR[0].z, bR[0].w};
        const float v1[4] = {bR[1].x, bR[1].y, bR[1].z, bR[1].w};
        #pragma unroll
        for (int j = 0; j < 4; ++j) {
            const int n = bCol4 + j;
            const int c = swz_blk(n, bRow2 >> 3) * 8 + (bRow2 & 7);
            union { __bf16 h[2]; unsigned int u; } p;
            p.h[0] = (__bf16)v0[j];
            p.h[1] = (__bf16)v1[j];
            *(unsigned int*)&Bt[n][c] = p.u;
        }
    };

    v8f acc00 = {}, acc01 = {}, acc10 = {}, acc11 = {};

    issue_loads(0);                         // prologue: first tile in flight

    for (int k0 = 0; k0 < K_; k0 += BK) {
        store_lds();                        // waits on in-flight loads (batched)
        __syncthreads();
        if (k0 + BK < K_)
            issue_loads(k0 + BK);           // next tile in flight across WMMAs

        // ---- A fragments: lane holds row M = lane%16 of its 16-row subtile.
        //      elems 0..7  : K = half*8 + 0..7      (block half)
        //      elems 8..15 : K = 16 + half*8 + 0..7 (block 2+half)
        v16bf afrag[2];
        #pragma unroll
        for (int ma = 0; ma < 2; ++ma) {
            const int row = wm * 32 + ma * 16 + r;
            const v8bf lo = *(const v8bf*)&As[row][swz_blk(row, half) * 8];
            const v8bf hi = *(const v8bf*)&As[row][swz_blk(row, 2 + half) * 8];
            afrag[ma] = __builtin_shufflevector(lo, hi,
                0, 1, 2, 3, 4, 5, 6, 7, 8, 9, 10, 11, 12, 13, 14, 15);
        }
        // ---- B fragments: lane holds col N = lane%16 of its 16-col subtile.
        //      elems 0..7  : K = half*8 + 0..7      (block half)
        //      elems 8..15 : K = 16 + half*8 + 0..7 (block 2+half)
        v16bf bfrag[2];
        #pragma unroll
        for (int nb = 0; nb < 2; ++nb) {
            const int n = wn * 32 + nb * 16 + r;
            const v8bf lo = *(const v8bf*)&Bt[n][swz_blk(n, half) * 8];
            const v8bf hi = *(const v8bf*)&Bt[n][swz_blk(n, 2 + half) * 8];
            bfrag[nb] = __builtin_shufflevector(lo, hi,
                0, 1, 2, 3, 4, 5, 6, 7, 8, 9, 10, 11, 12, 13, 14, 15);
        }

        acc00 = __builtin_amdgcn_wmma_f32_16x16x32_bf16(
                    false, afrag[0], false, bfrag[0], (short)0, acc00, false, false);
        acc01 = __builtin_amdgcn_wmma_f32_16x16x32_bf16(
                    false, afrag[0], false, bfrag[1], (short)0, acc01, false, false);
        acc10 = __builtin_amdgcn_wmma_f32_16x16x32_bf16(
                    false, afrag[1], false, bfrag[0], (short)0, acc10, false, false);
        acc11 = __builtin_amdgcn_wmma_f32_16x16x32_bf16(
                    false, afrag[1], false, bfrag[1], (short)0, acc11, false, false);
        __syncthreads();
    }

    // ---- epilogue: bias + sigmoid, route to ig / fg ----
    // C/D layout: VGPR j -> row (j + 8*half), col lane%16 within each 16x16.
    const int gcol0 = bn * BN + wn * 32 + r;        // nb = 0
    const int gcol1 = gcol0 + 16;                   // nb = 1
    const float b0 = bg[gcol0];
    const float b1 = bg[gcol1];
    v8f accs[2][2] = {{acc00, acc01}, {acc10, acc11}};
    #pragma unroll
    for (int ma = 0; ma < 2; ++ma) {
        #pragma unroll
        for (int j = 0; j < 8; ++j) {
            const int grow = bm * BM + wm * 32 + ma * 16 + j + 8 * half;
            const float v0 = sigmoidf_(accs[ma][0][j] + b0);
            const float v1 = sigmoidf_(accs[ma][1][j] + b1);
            if (gcol0 < D_) ig[(size_t)grow * D_ + gcol0] = v0;
            else            fg[(size_t)grow * D_ + (gcol0 - D_)] = v0;
            if (gcol1 < D_) ig[(size_t)grow * D_ + gcol1] = v1;
            else            fg[(size_t)grow * D_ + (gcol1 - D_)] = v1;
        }
    }
}

// ---------------------------------------------------------------------------
// Chunked parallel scan:  h_t = i_t*x_t + f_t*h_{t-1}
// ---------------------------------------------------------------------------
__global__ __launch_bounds__(256) void scan_phase1(
    const float* ig, const float* fg, const float* X,
    float* __restrict__ hpart, float* __restrict__ fprod)
{
    const int tid = blockIdx.x * blockDim.x + threadIdx.x;  // 0..B*D*NC-1
    const int d = tid % D_;
    const int c = (tid / D_) % NC;
    const int b = tid / (D_ * NC);

    float h = 0.0f, F = 1.0f;
    size_t idx = ((size_t)b * T_ + (size_t)c * CL) * D_ + d;
    for (int t = 0; t < CL; ++t, idx += D_) {
        const float i  = ig[idx];
        const float f  = fg[idx];
        const float xv = X[idx];
        h = fmaf(f, h, i * xv);
        F *= f;
    }
    const int ci = (b * D_ + d) * NC + c;
    hpart[ci] = h;
    fprod[ci] = F;
}

__global__ __launch_bounds__(256) void scan_phase2(
    const float* __restrict__ hpart, const float* __restrict__ fprod,
    float* __restrict__ hin)
{
    const int tid = blockIdx.x * blockDim.x + threadIdx.x;  // 0..B*D-1
    const int base = tid * NC;
    float h = 0.0f;
    #pragma unroll 4
    for (int c = 0; c < NC; ++c) {
        hin[base + c] = h;
        h = fmaf(fprod[base + c], h, hpart[base + c]);
    }
}

__global__ __launch_bounds__(256) void scan_phase3(
    const float* ig, const float* fg, const float* X,
    const float* __restrict__ hin, float* out)   // note: out aliases ig
{
    const int tid = blockIdx.x * blockDim.x + threadIdx.x;
    const int d = tid % D_;
    const int c = (tid / D_) % NC;
    const int b = tid / (D_ * NC);

    float h = hin[(b * D_ + d) * NC + c];
    size_t idx = ((size_t)b * T_ + (size_t)c * CL) * D_ + d;
    for (int t = 0; t < CL; ++t, idx += D_) {
        const float i  = ig[idx];
        const float f  = fg[idx];
        const float xv = X[idx];
        h = fmaf(f, h, i * xv);
        out[idx] = h;   // same index as reads above, owned by this thread
    }
}

// ---------------------------------------------------------------------------
extern "C" void kernel_launch(void* const* d_in, const int* in_sizes, int n_in,
                              void* d_out, int out_size, void* d_ws, size_t ws_size,
                              hipStream_t stream) {
    const float* X  = (const float*)d_in[0];   // [B,T,D]
    const float* Wg = (const float*)d_in[1];   // [D, 2D]
    const float* bg = (const float*)d_in[2];   // [2D]
    float* out = (float*)d_out;                // [B,T,D]

    // ig lives in d_out (overwritten in-place by phase3, index-owned per thread)
    float* ig = out;
    char*  ws = (char*)d_ws;
    float* fg    = (float*)ws;                                   // 64 MB
    float* hpart = (float*)(ws + (size_t)M_ * D_ * sizeof(float));
    float* fprod = hpart + (size_t)B_ * D_ * NC;
    float* hin   = fprod + (size_t)B_ * D_ * NC;

    dim3 gemm_grid(N_ / BN, M_ / BM);   // (32, 128)
    gates_gemm_kernel<<<gemm_grid, 256, 0, stream>>>(X, Wg, bg, ig, fg);

    const int n_chunk_threads = B_ * D_ * NC;   // 131072
    scan_phase1<<<n_chunk_threads / 256, 256, 0, stream>>>(ig, fg, X, hpart, fprod);
    scan_phase2<<<(B_ * D_) / 256, 256, 0, stream>>>(hpart, fprod, hin);
    scan_phase3<<<n_chunk_threads / 256, 256, 0, stream>>>(ig, fg, X, hin, out);
}
